// SS2D_22445499089450
// MI455X (gfx1250) — matrix operand order repaired and checked by reading, not compiled, not run on hardware
//
#include <hip/hip_runtime.h>
#include <hip/hip_bf16.h>
#include <stdint.h>

#define B0_   8
#define H_    96
#define W_    96
#define DM_   192
#define DI_   384
#define MTOK  (B0_*H_*W_)   /* 73728 token positions */
#define NO_   (2*DI_)       /* 768 = xf|z */

typedef __attribute__((ext_vector_type(16))) __bf16 v16bf;
typedef __attribute__((ext_vector_type(8)))  __bf16 v8bf;
typedef __attribute__((ext_vector_type(8)))  float  v8f;

static __device__ __forceinline__ __bf16 to_bf16(float f) {
  union { float f; uint32_t u; } v; v.f = f;
  uint32_t u = v.u;
  uint16_t h;
  if ((u & 0x7F800000u) == 0x7F800000u) {          // inf / NaN passthrough
    h = (uint16_t)(u >> 16);
    if (u & 0x007FFFFFu) h |= 0x0040;              // quiet NaN
  } else {
    h = (uint16_t)((u + 0x7FFFu + ((u >> 16) & 1u)) >> 16);  // RNE
  }
  union { uint16_t s; __bf16 b; } o; o.s = h; return o.b;
}

__global__ void cast_f32_bf16(const float* __restrict__ in, __bf16* __restrict__ out, int n) {
  for (int i = blockIdx.x * blockDim.x + threadIdx.x; i < n; i += gridDim.x * blockDim.x)
    out[i] = to_bf16(in[i]);
}

// C[M,N] = A[M,K] x B[N,K]^T.  A,B bf16 row-major (K contiguous), C fp32.
// One wave32 computes a 16(M) x 64(N) tile via v_wmma_f32_16x16x32_bf16.
// Requires M%16==0, N%64==0, K%32==0 (exact here: 73728, 768/192, 192/384).
__global__ __launch_bounds__(256) void gemm_bf16_wmma(
    const __bf16* __restrict__ A, const __bf16* __restrict__ B,
    float* __restrict__ C, int M, int N, int K) {
  const int lane   = threadIdx.x & 31;
  const int wave   = (blockIdx.x * blockDim.x + threadIdx.x) >> 5;
  const int nGrps  = N >> 6;
  const int mTile  = wave / nGrps;
  const int nBase  = (wave % nGrps) << 6;
  if (mTile * 16 >= M) return;                // wave-uniform: EXEC stays all-1s
  const int mlo = lane & 15;
  const int hi  = lane >> 4;

  const __bf16* aRow = A + (size_t)(mTile * 16 + mlo) * K;
  v8f acc0 = {}, acc1 = {}, acc2 = {}, acc3 = {};

  for (int k0 = 0; k0 < K; k0 += 32) {
    // A fragment: lane holds M=mlo, K = k0+hi*8+{0..7} and k0+16+hi*8+{0..7}
    v8bf alo = *(const v8bf*)(aRow + k0 + hi * 8);
    v8bf ahi = *(const v8bf*)(aRow + k0 + 16 + hi * 8);
    v16bf af;
#pragma unroll
    for (int t = 0; t < 8; ++t) { af[t] = alo[t]; af[8 + t] = ahi[t]; }

    // B fragment: lane holds N=n0+mlo, K = k0+hi*8+{0..7} and k0+16+hi*8+{0..7}
    const __bf16* bk = B + (size_t)(nBase + mlo) * K + k0 + hi * 8;
    v16bf b0 = __builtin_shufflevector(*(const v8bf*)(bk), *(const v8bf*)(bk + 16), 0,1,2,3,4,5,6,7,8,9,10,11,12,13,14,15);
    v16bf b1 = __builtin_shufflevector(*(const v8bf*)(bk + (size_t)16 * K), *(const v8bf*)(bk + (size_t)16 * K + 16), 0,1,2,3,4,5,6,7,8,9,10,11,12,13,14,15);
    v16bf b2 = __builtin_shufflevector(*(const v8bf*)(bk + (size_t)32 * K), *(const v8bf*)(bk + (size_t)32 * K + 16), 0,1,2,3,4,5,6,7,8,9,10,11,12,13,14,15);
    v16bf b3 = __builtin_shufflevector(*(const v8bf*)(bk + (size_t)48 * K), *(const v8bf*)(bk + (size_t)48 * K + 16), 0,1,2,3,4,5,6,7,8,9,10,11,12,13,14,15);

    acc0 = __builtin_amdgcn_wmma_f32_16x16x32_bf16(false, af, false, b0, (short)0, acc0, false, false);
    acc1 = __builtin_amdgcn_wmma_f32_16x16x32_bf16(false, af, false, b1, (short)0, acc1, false, false);
    acc2 = __builtin_amdgcn_wmma_f32_16x16x32_bf16(false, af, false, b2, (short)0, acc2, false, false);
    acc3 = __builtin_amdgcn_wmma_f32_16x16x32_bf16(false, af, false, b3, (short)0, acc3, false, false);
  }

  // C/D layout: element r -> (M = r + 8*hi, N = lane&15)
  float* cBase = C + (size_t)(mTile * 16 + hi * 8) * N + nBase + mlo;
#pragma unroll
  for (int r = 0; r < 8; ++r) {
    cBase[(size_t)r * N +  0] = acc0[r];
    cBase[(size_t)r * N + 16] = acc1[r];
    cBase[(size_t)r * N + 32] = acc2[r];
    cBase[(size_t)r * N + 48] = acc3[r];
  }
}

// Depthwise 7x7 conv, pad 3, on xf = xz[:, :, :, 0:384] (NHWC, row stride 768).
// Output xc NHWC (row stride 384). Threads vary c fastest -> coalesced.
__global__ __launch_bounds__(128) void dwconv7(
    const float* __restrict__ xz, const float* __restrict__ cw,
    const float* __restrict__ cb, float* __restrict__ xc) {
  const int c  = blockIdx.x * 128 + threadIdx.x;   // 0..383
  const int hw = blockIdx.y;                       // 0..9215
  const int b  = blockIdx.z;                       // 0..7
  const int h = hw / W_, w = hw % W_;
  float acc = cb[c];
  const float* wp = cw + c * 49;                   // conv_w[c,0,ky,kx]
#pragma unroll
  for (int ky = 0; ky < 7; ++ky) {
    const int y = h + ky - 3;
    if ((unsigned)y >= (unsigned)H_) continue;
#pragma unroll
    for (int kx = 0; kx < 7; ++kx) {
      const int x = w + kx - 3;
      if ((unsigned)x >= (unsigned)W_) continue;
      acc = fmaf(xz[(size_t)(b * (H_*W_) + y * W_ + x) * NO_ + c], wp[ky * 7 + kx], acc);
    }
  }
  xc[(size_t)(b * (H_*W_) + hw) * DI_ + c] = acc;
}

// Fused: 3x3 dilated box-sum (dilation = c/128 + 1, zero pad)  ->  LayerNorm(384)
//        -> + z residual (xz[:,384+c])  ->  bf16 cast for GEMM2.
// One block per pixel, 384 threads (one per channel), LDS reduction for mu/var.
__global__ __launch_bounds__(DI_) void boxsum_ln_bf16(
    const float* __restrict__ xc, const float* __restrict__ xz,
    const float* __restrict__ g, const float* __restrict__ bta,
    __bf16* __restrict__ Yb) {
  const int m  = blockIdx.x;       // token position
  const int c  = threadIdx.x;      // channel
  const int b  = m / (H_*W_);
  const int hw = m % (H_*W_);
  const int h  = hw / W_, w = hw % W_;
  const int d  = (c >> 7) + 1;     // dilation per 128-channel group

  float s = 0.f;
#pragma unroll
  for (int iy = -1; iy <= 1; ++iy) {
    const int y = h + iy * d;
    if ((unsigned)y >= (unsigned)H_) continue;
#pragma unroll
    for (int ix = -1; ix <= 1; ++ix) {
      const int x = w + ix * d;
      if ((unsigned)x >= (unsigned)W_) continue;
      s += xc[(size_t)(b * (H_*W_) + y * W_ + x) * DI_ + c];
    }
  }

  __shared__ float shs[DI_], shq[DI_];
  shs[c] = s; shq[c] = s * s;
  __syncthreads();
  if (c < 128) { shs[c] += shs[c + 128] + shs[c + 256];
                 shq[c] += shq[c + 128] + shq[c + 256]; }
  __syncthreads();
  for (int off = 64; off > 0; off >>= 1) {
    if (c < off) { shs[c] += shs[c + off]; shq[c] += shq[c + off]; }
    __syncthreads();
  }
  __shared__ float s_mu, s_inv;
  if (c == 0) {
    const float mu  = shs[0] * (1.f / DI_);
    const float var = shq[0] * (1.f / DI_) - mu * mu;
    s_mu = mu; s_inv = rsqrtf(var + 1e-5f);
  }
  __syncthreads();

  const float yv = (s - s_mu) * s_inv * g[c] + bta[c] + xz[(size_t)m * NO_ + DI_ + c];
  Yb[(size_t)m * DI_ + c] = to_bf16(yv);
}

extern "C" void kernel_launch(void* const* d_in, const int* in_sizes, int n_in,
                              void* d_out, int out_size, void* d_ws, size_t ws_size,
                              hipStream_t stream) {
  (void)in_sizes; (void)n_in; (void)out_size; (void)ws_size;
  const float* x      = (const float*)d_in[0];   // (8,96,96,192)
  const float* W_in   = (const float*)d_in[1];   // (768,192)
  const float* conv_w = (const float*)d_in[2];   // (384,1,7,7)
  const float* conv_b = (const float*)d_in[3];   // (384,)
  const float* ln_g   = (const float*)d_in[11];  // (384,)
  const float* ln_b   = (const float*)d_in[12];  // (384,)
  const float* W_out  = (const float*)d_in[13];  // (192,384)
  float* out = (float*)d_out;                    // (8,96,96,192) fp32

  // Workspace carve-out (256B-aligned); writes always precede reads.
  char* ws = (char*)d_ws;
  size_t off = 0;
  auto carve = [&](size_t bytes) -> char* {
    char* p = ws + off;
    off = (off + bytes + 255) & ~(size_t)255;
    return p;
  };
  __bf16* Xb = (__bf16*)carve((size_t)MTOK * DM_ * 2);  // x in bf16
  __bf16* Wi = (__bf16*)carve((size_t)NO_  * DM_ * 2);  // W_in bf16
  float*  XZ = (float*) carve((size_t)MTOK * NO_ * 4);  // xz = [xf | z]
  float*  XC = (float*) carve((size_t)MTOK * DI_ * 4);  // conv output (NHWC)
  __bf16* Yb = (__bf16*)carve((size_t)MTOK * DI_ * 2);  // LN output bf16
  __bf16* Wo = (__bf16*)carve((size_t)DM_  * DI_ * 2);  // W_out bf16

  // Stage operands in bf16 (range-safe; f32 accumulate in WMMA).
  cast_f32_bf16<<<2048, 256, 0, stream>>>(x,     Xb, MTOK * DM_);
  cast_f32_bf16<<<256,  256, 0, stream>>>(W_in,  Wi, NO_ * DM_);
  cast_f32_bf16<<<128,  256, 0, stream>>>(W_out, Wo, DM_ * DI_);

  // GEMM1: xz = x @ W_in^T   (73728 x 192) x (768 x 192)^T
  {
    const int waves = (MTOK / 16) * (NO_ / 64);   // 55296
    gemm_bf16_wmma<<<waves / 8, 256, 0, stream>>>(Xb, Wi, XZ, MTOK, NO_, DM_);
  }

  // Depthwise 7x7 conv on the xf half of xz.
  dwconv7<<<dim3(3, H_ * W_, B0_), 128, 0, stream>>>(XZ, conv_w, conv_b, XC);

  // A_M == softmax over a size-1 axis == 1, so attention reduces to a dilated
  // 3x3 box-sum of the conv output; fuse with LayerNorm + z residual + bf16.
  boxsum_ln_bf16<<<MTOK, DI_, 0, stream>>>(XC, XZ, ln_g, ln_b, Yb);

  // GEMM2: out = y @ W_out^T  (73728 x 384) x (192 x 384)^T
  {
    const int waves = (MTOK / 16) * (DM_ / 64);   // 13824
    gemm_bf16_wmma<<<waves / 8, 256, 0, stream>>>(Yb, Wo, out, MTOK, DM_, DI_);
  }
}
